// CrissCrossAttention_283467842727
// MI455X (gfx1250) — matrix operand order repaired and checked by reading, not compiled, not run on hardware
//
#include <hip/hip_runtime.h>

// ---------------- CDNA5 WMMA types ----------------
typedef __attribute__((ext_vector_type(16))) __bf16 v16bf;
typedef __attribute__((ext_vector_type(8)))  float  v8f;

#define WMMA_BF16(a,b,c) __builtin_amdgcn_wmma_f32_16x16x32_bf16(false,(a),false,(b),(short)0,(c),false,false)

#define NB 8
#define NC 256
#define NH 128
#define NW 128
#define ND 32
#define SPD 40    // q/k LDS row stride (halves): 80B, 16B-aligned, bank-spread
#define SAT 136   // att / V LDS row stride (halves): 272B
#define SX  264   // x / weight LDS row stride (halves): 528B

__device__ __forceinline__ unsigned short f2bf(float f) {
  unsigned int u = __float_as_uint(f);
  u += 0x7FFFu + ((u >> 16) & 1u);   // round-to-nearest-even to bf16
  return (unsigned short)(u >> 16);
}

// A operand (16x32 bf16, MxK): lane<16 holds k[kb..kb+8) & k[kb+16..kb+24); lane>=16: +8
__device__ __forceinline__ v16bf load_opA(const unsigned short* rowPtr, int kbase, int half) {
  uint4 lo = *(const uint4*)(rowPtr + kbase + half * 8);
  uint4 hi = *(const uint4*)(rowPtr + kbase + 16 + half * 8);
  v16bf r;
  __builtin_memcpy(&r, &lo, 16);
  __builtin_memcpy((char*)&r + 16, &hi, 16);
  return r;
}
// B operand (32x16 bf16, KxN): lane<16 holds k[kb..kb+8) & k[kb+16..kb+24); lane>=16: +8
__device__ __forceinline__ v16bf load_opB(const unsigned short* rowPtr, int kbase, int half) {
  uint4 lo = *(const uint4*)(rowPtr + kbase + half * 8);
  uint4 hi = *(const uint4*)(rowPtr + kbase + 16 + half * 8);
  v16bf r;
  __builtin_memcpy(&r, &lo, 16);
  __builtin_memcpy((char*)&r + 16, &hi, 16);
  return r;
}

// Async copy 16B global -> LDS (GVS mode), tracked by ASYNCcnt.
//   ldsOff : wave-relative LDS byte offset (flat LDS ptr truncated to 32b)
//   base   : uniform 64-bit global base (SGPR pair)
//   voff   : per-lane 32-bit byte offset
__device__ __forceinline__ void async_g2l_b128(unsigned ldsOff, const void* base, unsigned voff) {
  asm volatile("global_load_async_to_lds_b128 %0, %1, %2"
               :: "v"(ldsOff), "v"(voff), "s"(base) : "memory");
}
__device__ __forceinline__ void wait_asynccnt0() {
  asm volatile("s_wait_asynccnt 0" ::: "memory");
}

// =====================================================================
// Kernel 1: fused q/k/v projection. One WG per (b,h): all 320 output
// rows x 128 positions via bf16 WMMA (K=256).
// =====================================================================
__global__ __launch_bounds__(256) void cca_proj(
    const float* __restrict__ x,
    const float* __restrict__ Wq, const float* __restrict__ bq,
    const float* __restrict__ Wk, const float* __restrict__ bk,
    const float* __restrict__ Wv, const float* __restrict__ bv,
    unsigned short* __restrict__ qc,
    unsigned short* __restrict__ kc,
    unsigned short* __restrict__ vc)
{
  extern __shared__ unsigned short sm[];
  unsigned short* ldsX = sm;             // [128 pos][SX]  (B-operand layout: [n][k])
  unsigned short* ldsA = sm + NW * SX;   // [16 row][SX]   (A-operand layout: [m][k])

  const int tid = threadIdx.x;
  const int lane = tid & 31, wv = tid >> 5;
  const int half = lane >> 4, l16 = lane & 15;
  const int b = blockIdx.y, h = blockIdx.x;

  // Stage X[b, :, h, :] -> bf16 LDS transposed to [w][k] (coalesced global reads)
  const float* xbase = x + (size_t)b * NC * NH * NW + (size_t)h * NW;
  for (int it = tid; it < NW * (NC / 2); it += 256) {
    int w = it & (NW - 1);
    int kp = it >> 7;   // 0..127
    float f0 = xbase[(size_t)(2 * kp) * NH * NW + w];
    float f1 = xbase[(size_t)(2 * kp + 1) * NH * NW + w];
    unsigned int pk = (unsigned int)f2bf(f0) | ((unsigned int)f2bf(f1) << 16);
    *(unsigned int*)(ldsX + w * SX + 2 * kp) = pk;
  }
  __syncthreads();

  // B operands (X tile) are invariant across row-tiles: hoist all 8 K-steps.
  v16bf bx[8];
  #pragma unroll
  for (int kk = 0; kk < 8; ++kk)
    bx[kk] = load_opB(ldsX + (wv * 16 + l16) * SX, kk * 32, half);

  for (int rt = 0; rt < 20; ++rt) {        // 20 tiles of 16 output rows = 320 rows (q|k|v)
    for (int it = tid; it < 16 * (NC / 2); it += 256) {
      int j = it >> 7;
      int kp = it & 127;
      int r = rt * 16 + j;
      const float* wr;
      if (r < 32)      wr = Wq + (size_t)r * NC;
      else if (r < 64) wr = Wk + (size_t)(r - 32) * NC;
      else             wr = Wv + (size_t)(r - 64) * NC;
      unsigned int pk = (unsigned int)f2bf(wr[2 * kp]) | ((unsigned int)f2bf(wr[2 * kp + 1]) << 16);
      *(unsigned int*)(ldsA + j * SX + 2 * kp) = pk;
    }
    __syncthreads();

    v8f acc = {};
    #pragma unroll
    for (int kk = 0; kk < 8; ++kk) {
      v16bf a = load_opA(ldsA + l16 * SX, kk * 32, half);
      acc = WMMA_BF16(a, bx[kk], acc);
    }

    int w = wv * 16 + l16;
    int r0 = rt * 16 + half * 8;           // 8 contiguous output rows per lane
    unsigned short pk8[8];
    #pragma unroll
    for (int i = 0; i < 8; ++i) {
      int r = r0 + i;
      float bias = (r < 32) ? bq[r] : (r < 64) ? bk[r - 32] : bv[r - 64];
      pk8[i] = f2bf(acc[i] + bias);
    }
    uint4 st;
    __builtin_memcpy(&st, pk8, 16);
    if (rt < 2) {
      *(uint4*)(qc + ((size_t)(b * NW + w) * NH + h) * ND + r0) = st;
    } else if (rt < 4) {
      *(uint4*)(kc + ((size_t)(b * NW + w) * NH + h) * ND + (r0 - 32)) = st;
    } else {
      *(uint4*)(vc + ((size_t)(b * NW + w) * NH + h) * NC + (r0 - 64)) = st;
    }
    __syncthreads();
  }
}

// =====================================================================
// Kernel 2/3: per-direction softmax stats. One WG per (b,fixed).
// E = Q K^T (K=32, one WMMA per 16x16 tile); online rowwise max / sumexp.
// q/k staged via GLOBAL_LOAD_ASYNC_TO_LDS_B128 (ASYNCcnt path).
// =====================================================================
template <bool MASK>
__global__ __launch_bounds__(256) void cca_stats(
    const unsigned short* __restrict__ qb,
    const unsigned short* __restrict__ kbuf,
    float* __restrict__ mOut, float* __restrict__ sOut,
    int fA, int rA,      // q/k element strides (ushorts): base + fixed*fA + row*rA
    int rS, int fS)      // stats strides: b*H*W + row*rS + fixed*fS
{
  extern __shared__ unsigned short sm[];
  unsigned short* ldsQ = sm;             // [128][SPD]
  unsigned short* ldsK = sm + NH * SPD;  // [128][SPD]
  const int tid = threadIdx.x, lane = tid & 31, wv = tid >> 5;
  const int half = lane >> 4, l16 = lane & 15;
  const int b = blockIdx.y, fixed = blockIdx.x;
  const size_t qoff = (size_t)b * NH * NW * ND + (size_t)fixed * fA;

  // async-stage q/k: 128 rows x 64B, 16B chunks per lane, padded LDS scatter
  for (int it = tid; it < 128 * 4; it += 256) {
    int row = it >> 2, ch = it & 3;
    unsigned gByte = (unsigned)((qoff + (size_t)row * rA) * 2) + ch * 16;
    unsigned lByte = (unsigned)(size_t)(ldsQ + row * SPD) + ch * 16;
    async_g2l_b128(lByte, qb, gByte);
    lByte = (unsigned)(size_t)(ldsK + row * SPD) + ch * 16;
    async_g2l_b128(lByte, kbuf, gByte);
  }
  wait_asynccnt0();
  __syncthreads();

  float m[8], s[8];
  #pragma unroll
  for (int i = 0; i < 8; ++i) { m[i] = -3.4e38f; s[i] = 0.f; }

  v16bf a = load_opA(ldsQ + (wv * 16 + l16) * SPD, 0, half);
  for (int gt = 0; gt < 8; ++gt) {
    v16bf bb = load_opB(ldsK + (gt * 16 + l16) * SPD, 0, half);
    v8f e = {};
    e = WMMA_BF16(a, bb, e);
    int g = gt * 16 + l16;
    #pragma unroll
    for (int i = 0; i < 8; ++i) {
      int row = wv * 16 + i + half * 8;
      float ev = e[i];
      if (MASK && g == row) ev = -3.4e38f;                // diag -> -inf
      float tm = ev;
      tm = fmaxf(tm, __shfl_xor(tm, 1, 32));
      tm = fmaxf(tm, __shfl_xor(tm, 2, 32));
      tm = fmaxf(tm, __shfl_xor(tm, 4, 32));
      tm = fmaxf(tm, __shfl_xor(tm, 8, 32));
      float mn = fmaxf(m[i], tm);
      float p = __expf(ev - mn);
      float ts = p;
      ts += __shfl_xor(ts, 1, 32);
      ts += __shfl_xor(ts, 2, 32);
      ts += __shfl_xor(ts, 4, 32);
      ts += __shfl_xor(ts, 8, 32);
      s[i] = s[i] * __expf(m[i] - mn) + ts;
      m[i] = mn;
    }
  }
  if (l16 == 0) {
    size_t sbase = (size_t)b * NH * NW + (size_t)fixed * fS;
    #pragma unroll
    for (int i = 0; i < 8; ++i) {
      int row = wv * 16 + i + half * 8;
      mOut[sbase + (size_t)row * rS] = m[i];
      sOut[sbase + (size_t)row * rS] = s[i];
    }
  }
}

// =====================================================================
// Kernel 4/5: output GEMM. Recompute E, att = exp(E-m)/s (joint m,s),
// out = V @ att^T via WMMA (K=128). H-pass writes partial; W-pass fuses
// gamma*(outH+outW)+x into d_out.
// =====================================================================
template <bool MASK, bool FINAL>
__global__ __launch_bounds__(256) void cca_out(
    const unsigned short* __restrict__ qb,
    const unsigned short* __restrict__ kbuf,
    const unsigned short* __restrict__ vb,
    const float* __restrict__ mH, const float* __restrict__ sH,
    const float* __restrict__ mW, const float* __restrict__ sW,
    float* __restrict__ tmp,                // partial out_H: [b][w][c][h]
    const float* __restrict__ xin,
    const float* __restrict__ gamma,
    float* __restrict__ outp,
    int fA, int rA, int fV, int rV, int rS, int fS)
{
  extern __shared__ unsigned short sm[];
  unsigned short* ldsQ  = sm;                   // [128][SPD]
  unsigned short* ldsK  = ldsQ + NH * SPD;      // [128][SPD]
  unsigned short* ldsAT = ldsK + NH * SPD;      // [128][SAT] att, [row][g]
  unsigned short* ldsV  = ldsAT + NH * SAT;     // [256][SAT] V, [c][g]
  float* ldsM = (float*)(ldsV + NC * SAT);      // [128] joint max
  float* ldsR = ldsM + NH;                      // [128] 1/joint-sum

  const int tid = threadIdx.x, lane = tid & 31, wv = tid >> 5;
  const int half = lane >> 4, l16 = lane & 15;
  const int b = blockIdx.y, fixed = blockIdx.x;

  // async-stage q/k (latency hidden behind the manual V transpose below)
  const size_t qoff = (size_t)b * NH * NW * ND + (size_t)fixed * fA;
  for (int it = tid; it < 128 * 4; it += 256) {
    int row = it >> 2, ch = it & 3;
    unsigned gByte = (unsigned)((qoff + (size_t)row * rA) * 2) + ch * 16;
    unsigned lByte = (unsigned)(size_t)(ldsQ + row * SPD) + ch * 16;
    async_g2l_b128(lByte, qb, gByte);
    lByte = (unsigned)(size_t)(ldsK + row * SPD) + ch * 16;
    async_g2l_b128(lByte, kbuf, gByte);
  }
  // joint softmax stats for each row of this slice
  size_t sbase = (size_t)b * NH * NW + (size_t)fixed * fS;
  for (int row = tid; row < NH; row += 256) {
    size_t si = sbase + (size_t)row * rS;
    float m1 = mH[si], m2 = mW[si];
    float mm = fmaxf(m1, m2);
    float ss = sH[si] * __expf(m1 - mm) + sW[si] * __expf(m2 - mm);
    ldsM[row] = mm;
    ldsR[row] = 1.0f / ss;
  }
  // stage V slice transposed to [c][g] (transpose -> cannot use async copy)
  const size_t voff = (size_t)b * NH * NW * NC + (size_t)fixed * fV;
  for (int it = tid; it < 128 * 128; it += 256) {
    int g = it >> 7, cp = it & 127;
    unsigned int pk = *(const unsigned int*)(vb + voff + (size_t)g * rV + 2 * cp);
    ldsV[(2 * cp) * SAT + g]     = (unsigned short)(pk & 0xFFFFu);
    ldsV[(2 * cp + 1) * SAT + g] = (unsigned short)(pk >> 16);
  }
  wait_asynccnt0();
  __syncthreads();

  // recompute energies -> attention weights (bf16) in LDS
  {
    v16bf a = load_opA(ldsQ + (wv * 16 + l16) * SPD, 0, half);
    for (int gt = 0; gt < 8; ++gt) {
      v16bf bb = load_opB(ldsK + (gt * 16 + l16) * SPD, 0, half);
      v8f e = {};
      e = WMMA_BF16(a, bb, e);
      int g = gt * 16 + l16;
      #pragma unroll
      for (int i = 0; i < 8; ++i) {
        int row = wv * 16 + i + half * 8;
        float av = (MASK && g == row) ? 0.f
                 : __expf(e[i] - ldsM[row]) * ldsR[row];
        ldsAT[row * SAT + g] = f2bf(av);
      }
    }
  }
  __syncthreads();

  float g0 = FINAL ? gamma[0] : 0.f;
  // D[c, col] = sum_g V[c,g] * att[col,g]  (A = V tile, B = att rows)
  // B operands (att rows for this wave) are invariant across c-tiles: hoist.
  v16bf batt[4];
  #pragma unroll
  for (int kk = 0; kk < 4; ++kk)
    batt[kk] = load_opB(ldsAT + (wv * 16 + l16) * SAT, kk * 32, half);

  for (int ct = 0; ct < 16; ++ct) {
    v8f acc = {};
    #pragma unroll
    for (int kk = 0; kk < 4; ++kk) {
      v16bf a = load_opA(ldsV + (ct * 16 + l16) * SAT, kk * 32, half);
      acc = WMMA_BF16(a, batt[kk], acc);
    }
    int col = wv * 16 + l16;           // h' (H pass) or w' (W pass)
    int c0 = ct * 16 + half * 8;
    if (!FINAL) {
      float* dst = tmp + (((size_t)(b * NW + fixed) * NC + c0) * NH + col);
      #pragma unroll
      for (int i = 0; i < 8; ++i) dst[(size_t)i * NH] = acc[i];
    } else {
      #pragma unroll
      for (int i = 0; i < 8; ++i) {
        int c = c0 + i;
        float oh = tmp[((size_t)(b * NW + col) * NC + c) * NH + fixed];
        size_t xi = ((size_t)(b * NC + c) * NH + fixed) * NW + col;
        outp[xi] = g0 * (acc[i] + oh) + xin[xi];
      }
    }
  }
}

// =====================================================================
extern "C" void kernel_launch(void* const* d_in, const int* in_sizes, int n_in,
                              void* d_out, int out_size, void* d_ws, size_t ws_size,
                              hipStream_t stream)
{
  (void)in_sizes; (void)n_in; (void)out_size; (void)ws_size;
  const float* x  = (const float*)d_in[0];
  const float* Wq = (const float*)d_in[1];
  const float* bq = (const float*)d_in[2];
  const float* Wk = (const float*)d_in[3];
  const float* bk = (const float*)d_in[4];
  const float* Wv = (const float*)d_in[5];
  const float* bv = (const float*)d_in[6];
  const float* gm = (const float*)d_in[7];
  float* outp = (float*)d_out;

  unsigned char* ws = (unsigned char*)d_ws;
  size_t o = 0;
  auto alloc = [&](size_t bytes) -> void* {
    void* p = ws + o;
    o += (bytes + 255) & ~(size_t)255;
    return p;
  };
  unsigned short* qc = (unsigned short*)alloc((size_t)NB * NW * NH * ND * 2); // [b][w][h][d]
  unsigned short* kc = (unsigned short*)alloc((size_t)NB * NW * NH * ND * 2);
  unsigned short* vc = (unsigned short*)alloc((size_t)NB * NW * NH * NC * 2); // [b][w][h][c]
  float* mH   = (float*)alloc((size_t)NB * NH * NW * 4);
  float* sH   = (float*)alloc((size_t)NB * NH * NW * 4);
  float* mW   = (float*)alloc((size_t)NB * NH * NW * 4);
  float* sW   = (float*)alloc((size_t)NB * NH * NW * 4);
  float* tmpH = (float*)alloc((size_t)NB * NW * NC * NH * 4);                 // [b][w][c][h]

  dim3 grid(NW, NB), blk(256);

  size_t lds1 = (size_t)(NW * SX + 16 * SX) * 2;
  cca_proj<<<grid, blk, lds1, stream>>>(x, Wq, bq, Wk, bk, Wv, bv, qc, kc, vc);

  size_t lds2 = (size_t)(2 * NH * SPD) * 2;
  // H stats (fixed = w, row = h):  q elem stride fA=NH*ND, rA=ND ; stats rS=NW, fS=1
  cca_stats<true ><<<grid, blk, lds2, stream>>>(qc, kc, mH, sH, NH * ND, ND, NW, 1);
  // W stats (fixed = h, row = w)
  cca_stats<false><<<grid, blk, lds2, stream>>>(qc, kc, mW, sW, ND, NH * ND, 1, NW);

  size_t lds3 = (size_t)(2 * NH * SPD + NH * SAT + NC * SAT) * 2 + (size_t)2 * NH * 4;
  // H pass: fixed = w; V slice rows g=h -> fV=NH*NC, rV=NC. Writes partial out_H.
  cca_out<true, false><<<grid, blk, lds3, stream>>>(
      qc, kc, vc, mH, sH, mW, sW, tmpH, nullptr, nullptr, nullptr,
      NH * ND, ND, NH * NC, NC, NW, 1);
  // W pass: fixed = h; V slice rows g=w -> fV=NC, rV=NH*NC. Fuses final combine.
  cca_out<false, true><<<grid, blk, lds3, stream>>>(
      qc, kc, vc, mH, sH, mW, sW, tmpH, x, gm, outp,
      ND, NH * ND, NC, NH * NC, 1, NW);
}
